// s1_GAT_37074157699765
// MI455X (gfx1250) — matrix operand order repaired and checked by reading, not compiled, not run on hardware
//
#include <hip/hip_runtime.h>
#include <math.h>

// ---------------------------------------------------------------------------
// CDNA5 (gfx1250) GAT forward. wave32. Dense math via v_wmma_f32_16x16x32_f16.
// Edge passes restructured to per-node precompute (4B attention gathers,
// f32 z-row gathers, L2-resident working set).
// ---------------------------------------------------------------------------

typedef __attribute__((ext_vector_type(16))) _Float16 v16h;
typedef __attribute__((ext_vector_type(8)))  float    v8f;

#define NEG_SLOPE 0.2f
#define NORM_C    (1.0f / 20.0f)

__device__ __forceinline__ v8f wmma_f16(v16h a, v16h b, v8f c) {
  // D = A(16x32 f16) x B(32x16 f16) + C(16x16 f32)
  return __builtin_amdgcn_wmma_f32_16x16x32_f16(
      /*neg_a=*/false, a, /*neg_b=*/false, b,
      /*c_mod=*/(short)0, c, /*reuse_a=*/false, /*reuse_b=*/false);
}

// A-fragment (16x32 f16) from one f32 row of length 32.
// lane [0,15]  : M=lane,    K {0..15}
// lane [16,31] : M=lane-16, K {16..31}
__device__ __forceinline__ v16h frag_from_f32row(const float* __restrict__ row, int lane) {
  const int kb = (lane & 16) ? 16 : 0;
  const float4* p0 = (const float4*)(row + kb);
  const float4* p1 = (const float4*)(row + kb + 8);
  float4 f0 = p0[0], f1 = p0[1], f2 = p1[0], f3 = p1[1];
  v16h a;
  a[0]  = (_Float16)f0.x; a[1]  = (_Float16)f0.y; a[2]  = (_Float16)f0.z; a[3]  = (_Float16)f0.w;
  a[4]  = (_Float16)f1.x; a[5]  = (_Float16)f1.y; a[6]  = (_Float16)f1.z; a[7]  = (_Float16)f1.w;
  a[8]  = (_Float16)f2.x; a[9]  = (_Float16)f2.y; a[10] = (_Float16)f2.z; a[11] = (_Float16)f2.w;
  a[12] = (_Float16)f3.x; a[13] = (_Float16)f3.y; a[14] = (_Float16)f3.z; a[15] = (_Float16)f3.w;
  return a;
}

// Monotonic float <-> uint encoding for atomicMax-based float max reduction.
__device__ __forceinline__ unsigned enc_f(float v) {
  unsigned u = __float_as_uint(v);
  return (u & 0x80000000u) ? ~u : (u | 0x80000000u);
}
__device__ __forceinline__ float dec_f(unsigned k) {
  unsigned u = (k & 0x80000000u) ? (k & 0x7FFFFFFFu) : ~k;
  return __uint_as_float(u);
}

// --------------------------- per-layer prep --------------------------------
// W_self/W_neigh -> f16; vatt = W_neigh @ W_att; bsum = b_self + bias;
// scal = { c_att = b_neigh.W_att + b_att, ef_self = sum(W_edge) + b_edge }.
__global__ void prep_kernel(const float* __restrict__ Ws, const float* __restrict__ Wn,
                            const float* __restrict__ Wa, const float* __restrict__ We,
                            const float* __restrict__ ba, const float* __restrict__ be,
                            const float* __restrict__ bn, const float* __restrict__ bs,
                            const float* __restrict__ bias,
                            _Float16* __restrict__ Wsh, _Float16* __restrict__ Wnh,
                            float* __restrict__ vatt, float* __restrict__ bsum,
                            float* __restrict__ scal) {
  int tid = threadIdx.x;
  for (int j = tid; j < 1024; j += 256) {
    Wsh[j] = (_Float16)Ws[(j & 31) * 32 + (j >> 5)];
    Wnh[j] = (_Float16)Wn[(j & 31) * 32 + (j >> 5)];
  }
  if (tid < 32) {
    float acc = 0.f;
#pragma unroll
    for (int n = 0; n < 32; ++n) acc += Wn[tid * 32 + n] * Wa[n];
    vatt[tid] = acc;
    bsum[tid] = bs[tid] + bias[tid];
  }
  if (tid == 0) {
    float c = ba[0];
#pragma unroll
    for (int n = 0; n < 32; ++n) c += bn[n] * Wa[n];
    scal[0] = c;
    float e = be[0];
#pragma unroll
    for (int k = 0; k < 8; ++k) e += We[k];
    scal[1] = e;
  }
}

__global__ void init_red_kernel(unsigned* gmaxkey, float* gsum) {
  gmaxkey[0] = enc_f(-INFINITY);
  gsum[0] = 0.f;
}

// --------------------------- node GEMM (WMMA) ------------------------------
// y[N,32] = x[N,32] @ W(f16) + bias32.  Coalesced b128 stores via per-wave LDS.
__global__ void node_gemm_kernel(const float* __restrict__ x, const _Float16* __restrict__ Wh,
                                 const float* __restrict__ bias32, float* __restrict__ y,
                                 int N) {
  __shared__ float tile_all[8 * 512];
  int wv = threadIdx.x >> 5;
  int lane = threadIdx.x & 31;
  int wave = blockIdx.x * (blockDim.x >> 5) + wv;
  int ntiles = N >> 4;
  if (wave >= ntiles) return;
  int rowbase = wave << 4;
  int m = lane & 15;

  v16h a  = frag_from_f32row(x + (size_t)(rowbase + m) * 32, lane);
  v16h b0 = *(const v16h*)(Wh + m * 32 + (lane & 16));
  v16h b1 = *(const v16h*)(Wh + (m + 16) * 32 + (lane & 16));

  int col = m;
  float bc0 = bias32[col], bc1 = bias32[col + 16];
  v8f c0, c1;
#pragma unroll
  for (int r = 0; r < 8; ++r) { c0[r] = bc0; c1[r] = bc1; }
  c0 = wmma_f16(a, b0, c0);
  c1 = wmma_f16(a, b1, c1);

  // Stage 16x32 f32 tile in LDS, then coalesced float4 stores (2KB contiguous).
  float* hs = tile_all + wv * 512;
  int rb = (lane & 16) ? 8 : 0;
#pragma unroll
  for (int r = 0; r < 8; ++r) {
    hs[(rb + r) * 32 + col]      = c0[r];
    hs[(rb + r) * 32 + col + 16] = c1[r];
  }
  asm volatile("s_wait_dscnt 0x0" ::: "memory"); // same-wave DS RAW fence

  float4* dst = (float4*)(y + (size_t)rowbase * 32);
  const float4* srcv = (const float4*)hs;
#pragma unroll
  for (int it = 0; it < 4; ++it) dst[lane + 32 * it] = srcv[lane + 32 * it];
}

// --------------------------- node attention score --------------------------
// s[n] = leakyrelu( x[n] . vatt + c_att ).  Self-loops guarantee every node is
// a src, so softmax's global max over edges == max over s[n]: reduce it here.
__global__ void node_score_kernel(const float* __restrict__ x, const float* __restrict__ vatt,
                                  const float* __restrict__ scal, float* __restrict__ s,
                                  unsigned* __restrict__ gmaxkey, int N) {
  __shared__ float sv[32];
  __shared__ float sred[256];
  int tid = threadIdx.x;
  if (tid < 32) sv[tid] = vatt[tid];
  __syncthreads();

  int i = blockIdx.x * blockDim.x + tid;
  float aval = -INFINITY;
  if (i < N) {
    const float* row = x + (size_t)i * 32;
    float d = scal[0];
#pragma unroll
    for (int k = 0; k < 32; ++k) d += row[k] * sv[k];
    float a = (d >= 0.f) ? d : NEG_SLOPE * d;
    s[i] = a;
    aval = a;
  }
  sred[tid] = aval;
  __syncthreads();
  for (int st = 128; st > 0; st >>= 1) {
    if (tid < st) sred[tid] = fmaxf(sred[tid], sred[tid + st]);
    __syncthreads();
  }
  if (tid == 0) atomicMax(gmaxkey, enc_f(sred[0]));
}

// --------------------------- edge pass -------------------------------------
// One pass: att[i] = exp(s[src_i] - gmax) (4B gather), ef[i] = ea_i.W_edge + b_edge,
// block-reduced sum of exp -> gsum.
__global__ void edge_pass_kernel(const int* __restrict__ ei, const float* __restrict__ ea,
                                 const float* __restrict__ s, const float* __restrict__ scal,
                                 const float* __restrict__ We, const float* __restrict__ be,
                                 const unsigned* __restrict__ gmaxkey,
                                 float* __restrict__ att, float* __restrict__ ef,
                                 float* __restrict__ gsum, int E, int N) {
  __shared__ float swe[8];
  __shared__ float sred[256];
  int tid = threadIdx.x;
  if (tid < 8) swe[tid] = We[tid];
  __syncthreads();

  int i = blockIdx.x * blockDim.x + tid;
  int total = E + N;
  float gmax = dec_f(gmaxkey[0]);
  float v = 0.f;
  if (i < total) {
    int src = (i < E) ? ei[i] : (i - E);
    float e = expf(s[src] - gmax);
    att[i] = e;
    v = e;
    float f;
    if (i < E) {
      const float4* ep = (const float4*)(ea + (size_t)i * 8);
      float4 u0 = ep[0], u1 = ep[1];
      f = be[0] + u0.x * swe[0] + u0.y * swe[1] + u0.z * swe[2] + u0.w * swe[3]
                + u1.x * swe[4] + u1.y * swe[5] + u1.z * swe[6] + u1.w * swe[7];
    } else {
      f = scal[1];
    }
    ef[i] = f;
  }
  sred[tid] = v;
  __syncthreads();
  for (int st = 128; st > 0; st >>= 1) {
    if (tid < st) sred[tid] += sred[tid + st];
    __syncthreads();
  }
  if (tid == 0) atomicAdd(gsum, sred[0]);
}

// --------------------------- message scatter -------------------------------
// msg_i = z[src_i] * (att_i/gsum) + ef_i*NORM_C, scatter-add to y[dst_i].
// 8 consecutive threads cover one 128B z-row (coalesced); 4 f32 atomics each.
__global__ void msg_scatter_kernel(const float* __restrict__ z, const int* __restrict__ ei,
                                   const float* __restrict__ att, const float* __restrict__ ef,
                                   const float* __restrict__ gsum, float* __restrict__ y,
                                   int E, int N) {
  long long t = (long long)blockIdx.x * blockDim.x + threadIdx.x;
  int total = E + N;
  if (t >= (long long)total * 8) return;
  int i = (int)(t >> 3);
  int c4 = (int)(t & 7);

  int src = (i < E) ? ei[i] : (i - E);
  int dst = (i < E) ? ei[E + i] : (i - E);
  float w = att[i] * (1.0f / gsum[0]);
  float add = ef[i] * NORM_C;

  float4 zc = ((const float4*)(z + (size_t)src * 32))[c4];
  float* yp = y + (size_t)dst * 32 + c4 * 4;
  atomicAdd(yp + 0, zc.x * w + add);
  atomicAdd(yp + 1, zc.y * w + add);
  atomicAdd(yp + 2, zc.z * w + add);
  atomicAdd(yp + 3, zc.w * w + add);
}

// --------------------------- elementwise ReLU ------------------------------
__global__ void relu_kernel(float* __restrict__ y, int count) {
  int i = blockIdx.x * blockDim.x + threadIdx.x;
  if (i < count) y[i] = fmaxf(y[i], 0.f);
}

// --------------------------- FC head prep ----------------------------------
__global__ void fcprep_kernel(const float* __restrict__ W1, _Float16* __restrict__ W1h) {
  int tid = threadIdx.x;
  for (int j = tid; j < 2048; j += 256) W1h[j] = (_Float16)W1[(j & 63) * 32 + (j >> 6)];
}

// --------------------------- FC head (WMMA) --------------------------------
// out[p,:] = relu(concat(xl[l_index[p]], xr[r_index[p]]) @ W_fc1 + b_fc1) @ W_fc2 + b_fc2
__global__ void fc_kernel(const float* __restrict__ xl, const float* __restrict__ xr,
                          const int* __restrict__ lidx, const int* __restrict__ ridx,
                          const _Float16* __restrict__ W1h, const float* __restrict__ b1,
                          const float* __restrict__ W2, const float* __restrict__ b2,
                          float* __restrict__ out, int P) {
  __shared__ float hs_all[8 * 512]; // per-wave 16x32 hidden tile
  int wv = threadIdx.x >> 5;
  int lane = threadIdx.x & 31;
  int wave = blockIdx.x * (blockDim.x >> 5) + wv;
  int ntiles = P >> 4;
  if (wave >= ntiles) return;
  int base = wave << 4;
  int m = lane & 15;
  int p = base + m;
  int il = lidx[p], ir = ridx[p];

  v16h a0 = frag_from_f32row(xl + (size_t)il * 32, lane); // K = 0..31  (left)
  v16h a1 = frag_from_f32row(xr + (size_t)ir * 32, lane); // K = 32..63 (right)
  v16h b00 = *(const v16h*)(W1h + m * 64 + (lane & 16));
  v16h b01 = *(const v16h*)(W1h + (m + 16) * 64 + (lane & 16));
  v16h b10 = *(const v16h*)(W1h + m * 64 + 32 + (lane & 16));
  v16h b11 = *(const v16h*)(W1h + (m + 16) * 64 + 32 + (lane & 16));

  int col = m;
  float bc0 = b1[col], bc1 = b1[col + 16];
  v8f c0, c1;
#pragma unroll
  for (int r = 0; r < 8; ++r) { c0[r] = bc0; c1[r] = bc1; }
  c0 = wmma_f16(a1, b10, c0);
  c0 = wmma_f16(a0, b00, c0);
  c1 = wmma_f16(a1, b11, c1);
  c1 = wmma_f16(a0, b01, c1);

  float* hs = hs_all + wv * 512;
  int rb = (lane & 16) ? 8 : 0;
#pragma unroll
  for (int r = 0; r < 8; ++r) {
    hs[(rb + r) * 32 + col]      = fmaxf(c0[r], 0.f);
    hs[(rb + r) * 32 + col + 16] = fmaxf(c1[r], 0.f);
  }
  asm volatile("s_wait_dscnt 0x0" ::: "memory"); // same-wave DS RAW fence

  int m2 = lane >> 1;
  int o = lane & 1;
  float acc = b2[o];
#pragma unroll
  for (int n = 0; n < 32; ++n) acc += hs[m2 * 32 + n] * W2[n * 2 + o];
  out[(size_t)(base + m2) * 2 + o] = acc;
}

// ===========================================================================
extern "C" void kernel_launch(void* const* d_in, const int* in_sizes, int n_in,
                              void* d_out, int out_size, void* d_ws, size_t ws_size,
                              hipStream_t stream) {
  const float* x_l  = (const float*)d_in[0];
  const int*   ei_l = (const int*)d_in[1];
  const float* ea_l = (const float*)d_in[2];
  const float* x_r  = (const float*)d_in[3];
  const int*   ei_r = (const int*)d_in[4];
  const float* ea_r = (const float*)d_in[5];
  const int*   lidx = (const int*)d_in[6];
  const int*   ridx = (const int*)d_in[7];
  // params pytree, dict keys sorted: W_fc1, W_fc2, b_fc1, b_fc2, gat1_l, gat1_r, gat2_l, gat2_r
  const float* W_fc1 = (const float*)d_in[8];
  const float* W_fc2 = (const float*)d_in[9];
  const float* b_fc1 = (const float*)d_in[10];
  const float* b_fc2 = (const float*)d_in[11];

  const int N = in_sizes[0] / 32;
  const int E = in_sizes[1] / 2;
  const int P = in_sizes[6];
  const int TOT = E + N;

  // workspace layout
  char* ws = (char*)d_ws;
  size_t off = 0;
  auto carve = [&](size_t bytes) -> char* {
    char* p = ws + off;
    off = (off + bytes + 255) & ~(size_t)255;
    return p;
  };
  float*     bufA    = (float*)carve((size_t)N * 32 * 4);
  float*     bufB    = (float*)carve((size_t)N * 32 * 4);
  float*     bufC    = (float*)carve((size_t)N * 32 * 4);
  float*     zbuf    = (float*)carve((size_t)N * 32 * 4);  // neigh-transformed nodes
  float*     snode   = (float*)carve((size_t)N * 4);       // per-node attention score
  float*     att     = (float*)carve((size_t)TOT * 4);
  float*     ef      = (float*)carve((size_t)TOT * 4);
  _Float16*  Wsh     = (_Float16*)carve(1024 * 2);
  _Float16*  Wnh     = (_Float16*)carve(1024 * 2);
  float*     vatt    = (float*)carve(32 * 4);
  float*     bsum    = (float*)carve(32 * 4);
  float*     scal    = (float*)carve(64);
  unsigned*  gmaxkey = (unsigned*)carve(64);
  float*     gsum    = (float*)carve(64);
  _Float16*  W1h     = (_Float16*)carve(2048 * 2);

  struct Cfg { const float* x; const int* ei; const float* ea; int pb; float* y; int relu; };
  // pytree order inside each gat dict (sorted keys):
  // +0 W_att, +1 W_edge, +2 W_neigh, +3 W_self, +4 b_att, +5 b_edge, +6 b_neigh, +7 b_self, +8 bias
  Cfg cfgs[4] = {
      { x_l,  ei_l, ea_l, 12, bufA, 1 },  // gat1_l
      { bufA, ei_l, ea_l, 30, bufB, 0 },  // gat2_l
      { x_r,  ei_r, ea_r, 21, bufA, 1 },  // gat1_r
      { bufA, ei_r, ea_r, 39, bufC, 0 },  // gat2_r
  };

  const int node_tiles = N / 16;

  for (int L = 0; L < 4; ++L) {
    const Cfg& c = cfgs[L];
    const float* Wa = (const float*)d_in[c.pb + 0];
    const float* We = (const float*)d_in[c.pb + 1];
    const float* Wn = (const float*)d_in[c.pb + 2];
    const float* Ws = (const float*)d_in[c.pb + 3];
    const float* ba = (const float*)d_in[c.pb + 4];
    const float* be = (const float*)d_in[c.pb + 5];
    const float* bn = (const float*)d_in[c.pb + 6];
    const float* bs = (const float*)d_in[c.pb + 7];
    const float* bi = (const float*)d_in[c.pb + 8];

    prep_kernel<<<1, 256, 0, stream>>>(Ws, Wn, Wa, We, ba, be, bn, bs, bi,
                                       Wsh, Wnh, vatt, bsum, scal);
    init_red_kernel<<<1, 1, 0, stream>>>(gmaxkey, gsum);

    // y = x@W_self + (b_self + bias);  z = x@W_neigh + b_neigh  (both WMMA)
    node_gemm_kernel<<<(node_tiles + 7) / 8, 256, 0, stream>>>(c.x, Wsh, bsum, c.y, N);
    node_gemm_kernel<<<(node_tiles + 7) / 8, 256, 0, stream>>>(c.x, Wnh, bn, zbuf, N);

    node_score_kernel<<<(N + 255) / 256, 256, 0, stream>>>(c.x, vatt, scal, snode,
                                                           gmaxkey, N);
    edge_pass_kernel<<<(TOT + 255) / 256, 256, 0, stream>>>(c.ei, c.ea, snode, scal,
                                                            We, be, gmaxkey, att, ef,
                                                            gsum, E, N);
    long long sthreads = (long long)TOT * 8;
    msg_scatter_kernel<<<(unsigned)((sthreads + 255) / 256), 256, 0, stream>>>(
        zbuf, c.ei, att, ef, gsum, c.y, E, N);

    if (c.relu)
      relu_kernel<<<((size_t)N * 32 + 255) / 256, 256, 0, stream>>>(c.y, N * 32);
  }

  fcprep_kernel<<<1, 256, 0, stream>>>(W_fc1, W1h);
  fc_kernel<<<((P / 16) + 7) / 8, 256, 0, stream>>>(bufB, bufC, lidx, ridx, W1h, b_fc1,
                                                    W_fc2, b_fc2, (float*)d_out, P);
}
